// DilatedOCA_40467181862932
// MI455X (gfx1250) — matrix operand order repaired and checked by reading, not compiled, not run on hardware
//
#include <hip/hip_runtime.h>
#include <hip/hip_bf16.h>
#include <cstddef>

// ---------------- problem constants ----------------
#define HW      65536      // 256*256
#define IMG     256
#define NWIN    32         // 256/8
#define WSZ     8
#define OWSZ    12
#define NTOK    64         // WSZ*WSZ
#define MTOK    144        // OWSZ*OWSZ
#define HEADS   4
#define DH      32
#define INNER   128
#define SCALE_F 0.17677669529663689f   // 32^-0.5
#define NEG_INF (-3.0e38f)

typedef __attribute__((ext_vector_type(16))) __bf16 v16bf;
typedef __attribute__((ext_vector_type(8)))  __bf16 v8bf;
typedef __attribute__((ext_vector_type(8)))  float  v8f;
typedef __attribute__((ext_vector_type(4)))  float  v4f;

__device__ __forceinline__ __bf16 f2bf(float f)  { return static_cast<__bf16>(f); }
__device__ __forceinline__ float  bf2f(__bf16 h) { return static_cast<float>(h); }

__device__ __forceinline__ v16bf concat8(v8bf lo, v8bf hi) {
    return __builtin_shufflevector(lo, hi, 0,1,2,3,4,5,6,7,8,9,10,11,12,13,14,15);
}
__device__ __forceinline__ v8f wmma_bf16(v16bf a, v16bf b, v8f c) {
    return __builtin_amdgcn_wmma_f32_16x16x32_bf16(
        false, a, false, b, (short)0, c, false, false);
}
// A-operand fragment: lane owns row M; elements = 16 contiguous K from k0
__device__ __forceinline__ v16bf loadA(const __bf16* row, int k0) {
    v8bf lo = *(const v8bf*)(row + k0);
    v8bf hi = *(const v8bf*)(row + k0 + 8);
    return concat8(lo, hi);
}
// B-operand fragment: lane owns column N; elements = 16 contiguous K from k0
__device__ __forceinline__ v16bf loadB(const __bf16* row, int k0) {
    v8bf lo = *(const v8bf*)(row + k0);
    v8bf hi = *(const v8bf*)(row + k0 + 8);
    return concat8(lo, hi);
}
__device__ __forceinline__ void storeC(float* dst, v8f c) {   // 8 contiguous f32
    *(v4f*)dst       = __builtin_shufflevector(c, c, 0, 1, 2, 3);
    *(v4f*)(dst + 4) = __builtin_shufflevector(c, c, 4, 5, 6, 7);
}

// =====================================================================
// Kernel 1: qkv[b][o][p] = sum_c Wqkv[o][c] * x[b][c][p]   (o in [0,384))
// grid (1024, 3, 2), block 256 (8 waves). Tile: M=128, N=64, K=128.
// =====================================================================
__global__ __launch_bounds__(256) void qkv_gemm(
    const float* __restrict__ x, const float* __restrict__ Wqkv,
    __bf16* __restrict__ qkv)
{
    __shared__ __bf16 aW [128][136];   // A: [o][c]
    __shared__ __bf16 bXT[ 64][136];   // B transposed: [pixel][c]
    __shared__ __bf16 oS [128][ 72];   // epilogue staging: [o][pixel]

    const int tid = threadIdx.x;
    const int p0  = blockIdx.x * 64;
    const int mb  = blockIdx.y;
    const int bz  = blockIdx.z;

    for (int idx = tid; idx < 128 * 32; idx += 256) {
        int i = idx >> 5, c4 = (idx & 31) * 4;
        v4f w = *(const v4f*)(Wqkv + (size_t)(mb * 128 + i) * 128 + c4);
        #pragma unroll
        for (int j = 0; j < 4; ++j) {
            float f = w[j];
            if (mb == 0) f *= SCALE_F;           // fold q-scale into W_q
            aW[i][c4 + j] = f2bf(f);
        }
    }
    for (int idx = tid; idx < 128 * 16; idx += 256) {
        int c = idx >> 4, j4 = (idx & 15) * 4;
        v4f xv = *(const v4f*)(x + ((size_t)bz * 128 + c) * HW + p0 + j4);
        #pragma unroll
        for (int j = 0; j < 4; ++j) bXT[j4 + j][c] = f2bf(xv[j]);
    }
    __syncthreads();

    const int wv = tid >> 5, lane = tid & 31;
    const int lm = lane & 15, hi = lane >> 4;

    v8f acc[4] = {};
    #pragma unroll
    for (int kk = 0; kk < 4; ++kk) {
        v16bf a = loadA(&aW[wv * 16 + lm][0], kk * 32 + hi * 16);
        #pragma unroll
        for (int nt = 0; nt < 4; ++nt) {
            v16bf b = loadB(&bXT[nt * 16 + lm][0], kk * 32 + hi * 16);
            acc[nt] = wmma_bf16(a, b, acc[nt]);
        }
    }

    #pragma unroll
    for (int nt = 0; nt < 4; ++nt)
        #pragma unroll
        for (int g = 0; g < 8; ++g)
            oS[wv * 16 + hi * 8 + g][nt * 16 + lm] = f2bf(acc[nt][g]);
    __syncthreads();
    for (int idx = tid; idx < 128 * 8; idx += 256) {
        int i = idx >> 3, j8 = (idx & 7) * 8;
        v8bf vv = *(const v8bf*)&oS[i][j8];
        *(v8bf*)(qkv + ((size_t)bz * 384 + mb * 128 + i) * HW + p0 + j8) = vv;
    }
}

// =====================================================================
// Kernel 2: attention, one block (128 thr, 4 waves) per (window, head).
// S=QK^T, rel-logits (RW=Q.relW^T, RH=Q.relH^T) and O=PV all via WMMA.
// =====================================================================
__global__ __launch_bounds__(128) void attn_kernel(
    const __bf16* __restrict__ qkv,
    const float* __restrict__ relH, const float* __restrict__ relW,
    __bf16* __restrict__ ob)
{
    __shared__ __bf16 q_s [NTOK][40];       // A of QK^T / rel: [n][d]
    __shared__ __bf16 k_s [MTOK][40];       // B of QK^T: [m][d]
    __shared__ __bf16 relB_w[32][40];       // B of RW: [r][d], rows 23..31 zero
    __shared__ __bf16 relB_h[32][40];       // B of RH: [r][d], rows 23..31 zero
    __shared__ __bf16 vT  [DH ][168];       // B of PV (transposed): [d][m], m-pad zero
    __shared__ float  sT  [MTOK][68];       // scores transposed: [m][n]
    __shared__ float  rwT [32][68];         // RW transposed: [r][n]
    __shared__ float  rhT [32][68];         // RH transposed: [r][n]
    __shared__ __bf16 p_s [NTOK][168];      // A of PV: [n][m], m-pad zero
    __shared__ __bf16 oT  [DH ][72];        // output staging: [d][n]

    const int bx   = blockIdx.x;
    const int head = bx & 3;
    const int widx = bx >> 2;
    const int b    = widx >> 10;
    const int wr   = (widx >> 5) & 31;
    const int wc   = widx & 31;
    const int tid  = threadIdx.x;

    const size_t qbase = ((size_t)b * 384 +       head * DH) * HW;
    const size_t kbase = ((size_t)b * 384 + 128 + head * DH) * HW;
    const size_t vbase = ((size_t)b * 384 + 256 + head * DH) * HW;
    const __bf16 BZ = f2bf(0.0f);

    for (int idx = tid; idx < DH * 8; idx += 128) {
        int d = idx >> 3, rr = idx & 7;
        v8bf g = *(const v8bf*)(qkv + qbase + (size_t)d * HW + (wr * 8 + rr) * IMG + wc * 8);
        #pragma unroll
        for (int j = 0; j < 8; ++j) q_s[rr * 8 + j][d] = g[j];
    }
    for (int idx = tid; idx < MTOK * DH; idx += 128) {
        int m = idx % MTOK, d = idx / MTOK;
        int r = wr * WSZ - 2 + m / OWSZ, c = wc * WSZ - 2 + m % OWSZ;
        bool ok = (r >= 0) && (r < IMG) && (c >= 0) && (c < IMG);
        size_t off = (size_t)d * HW + r * IMG + c;
        k_s[m][d] = ok ? qkv[kbase + off] : BZ;
        vT[d][m]  = ok ? qkv[vbase + off] : BZ;
    }
    for (int idx = tid; idx < DH * 16; idx += 128)          // zero K-pad of V
        vT[idx >> 4][MTOK + (idx & 15)] = BZ;
    for (int idx = tid; idx < NTOK * 16; idx += 128)        // zero K-pad of P
        p_s[idx >> 4][MTOK + (idx & 15)] = BZ;
    for (int idx = tid; idx < 32 * DH; idx += 128) {        // rel tables -> bf16, r-pad zero
        int r = idx >> 5, d = idx & 31;
        relB_w[r][d] = (r < 23) ? f2bf(relW[idx]) : BZ;
        relB_h[r][d] = (r < 23) ? f2bf(relH[idx]) : BZ;
    }
    __syncthreads();

    const int wv = tid >> 5, lane = tid & 31;
    const int lm = lane & 15, hi = lane >> 4;
    const int n0 = wv * 16 + hi * 8;

    // ---- S = Q K^T  and  RW/RH = Q rel^T (same A fragment) ----
    {
        v16bf a = loadA(&q_s[wv * 16 + lm][0], hi * 16);
        for (int mt = 0; mt < 9; ++mt) {
            v16bf bq = loadB(&k_s[mt * 16 + lm][0], hi * 16);
            v8f c = {};
            c = wmma_bf16(a, bq, c);
            storeC(&sT[mt * 16 + lm][n0], c);
        }
        #pragma unroll
        for (int rt = 0; rt < 2; ++rt) {
            v16bf bw = loadB(&relB_w[rt * 16 + lm][0], hi * 16);
            v8f cw = {};
            cw = wmma_bf16(a, bw, cw);
            storeC(&rwT[rt * 16 + lm][n0], cw);
            v16bf bh = loadB(&relB_h[rt * 16 + lm][0], hi * 16);
            v8f ch = {};
            ch = wmma_bf16(a, bh, ch);
            storeC(&rhT[rt * 16 + lm][n0], ch);
        }
    }
    __syncthreads();

    // ---- mask + softmax : one row per lane (reads coalesced over n) ----
    if (tid < NTOK) {
        int n = tid, qr = n >> 3, qc = n & 7;
        float rw[OWSZ], rh[OWSZ];
        #pragma unroll
        for (int j = 0; j < OWSZ; ++j) {
            rw[j] = rwT[11 + j - qc][n];
            rh[j] = rhT[11 + j - qr][n];
        }
        float mx = NEG_INF;
        for (int m = 0; m < MTOK; ++m) {
            int kr = m / OWSZ, kc = m % OWSZ;
            float sv = sT[m][n] + rw[kc] + rh[kr];
            if (((qr ^ kr) & 1) | ((qc ^ kc) & 1)) sv = NEG_INF;
            sT[m][n] = sv;
            mx = fmaxf(mx, sv);
        }
        float sum = 0.f;
        for (int m = 0; m < MTOK; ++m) {
            float e = __expf(sT[m][n] - mx);
            sT[m][n] = e;
            sum += e;
        }
        float inv = 1.0f / sum;
        for (int m = 0; m < MTOK; ++m)
            p_s[n][m] = f2bf(sT[m][n] * inv);
    }
    __syncthreads();

    // ---- O = P V : K = 144 padded to 160 (5 WMMA k-steps) ----
    {
        v8f acc[2] = {};
        #pragma unroll
        for (int kk = 0; kk < 5; ++kk) {
            v16bf a = loadA(&p_s[wv * 16 + lm][0], kk * 32 + hi * 16);
            #pragma unroll
            for (int dt = 0; dt < 2; ++dt) {
                v16bf bv = loadB(&vT[dt * 16 + lm][0], kk * 32 + hi * 16);
                acc[dt] = wmma_bf16(a, bv, acc[dt]);
            }
        }
        #pragma unroll
        for (int dt = 0; dt < 2; ++dt) {
            v8bf o8;
            #pragma unroll
            for (int g = 0; g < 8; ++g) o8[g] = f2bf(acc[dt][g]);
            *(v8bf*)&oT[dt * 16 + lm][n0] = o8;
        }
    }
    __syncthreads();

    const size_t obase = ((size_t)b * INNER + head * DH) * HW;
    for (int idx = tid; idx < DH * 8; idx += 128) {
        int d = idx >> 3, rr = idx & 7;
        v8bf vv = *(const v8bf*)&oT[d][rr * 8];
        *(v8bf*)(ob + obase + (size_t)d * HW + (wr * 8 + rr) * IMG + wc * 8) = vv;
    }
}

// =====================================================================
// Kernel 3: out[b][o][p] = sum_c Wout[o][c] * ob[b][c][p]  (fp32 output)
// grid (1024, 1, 2), block 256.
// =====================================================================
__global__ __launch_bounds__(256) void out_gemm(
    const __bf16* __restrict__ ob, const float* __restrict__ Wout,
    float* __restrict__ out)
{
    __shared__ __bf16 aW [128][136];
    __shared__ __bf16 bXT[ 64][136];
    __shared__ float  oSf[128][ 68];

    const int tid = threadIdx.x;
    const int p0  = blockIdx.x * 64;
    const int bz  = blockIdx.z;

    for (int idx = tid; idx < 128 * 32; idx += 256) {
        int i = idx >> 5, c4 = (idx & 31) * 4;
        v4f w = *(const v4f*)(Wout + (size_t)i * 128 + c4);
        #pragma unroll
        for (int j = 0; j < 4; ++j) aW[i][c4 + j] = f2bf(w[j]);
    }
    for (int idx = tid; idx < 128 * 8; idx += 256) {
        int c = idx >> 3, j8 = (idx & 7) * 8;
        v8bf g = *(const v8bf*)(ob + ((size_t)bz * 128 + c) * HW + p0 + j8);
        #pragma unroll
        for (int j = 0; j < 8; ++j) bXT[j8 + j][c] = g[j];
    }
    __syncthreads();

    const int wv = tid >> 5, lane = tid & 31;
    const int lm = lane & 15, hi = lane >> 4;

    v8f acc[4] = {};
    #pragma unroll
    for (int kk = 0; kk < 4; ++kk) {
        v16bf a = loadA(&aW[wv * 16 + lm][0], kk * 32 + hi * 16);
        #pragma unroll
        for (int nt = 0; nt < 4; ++nt) {
            v16bf b = loadB(&bXT[nt * 16 + lm][0], kk * 32 + hi * 16);
            acc[nt] = wmma_bf16(a, b, acc[nt]);
        }
    }

    #pragma unroll
    for (int nt = 0; nt < 4; ++nt)
        #pragma unroll
        for (int g = 0; g < 8; ++g)
            oSf[wv * 16 + hi * 8 + g][nt * 16 + lm] = acc[nt][g];
    __syncthreads();
    for (int idx = tid; idx < 128 * 16; idx += 256) {
        int i = idx >> 4, j4 = (idx & 15) * 4;
        v4f vv = *(const v4f*)&oSf[i][j4];
        *(v4f*)(out + ((size_t)bz * 128 + i) * HW + p0 + j4) = vv;
    }
}

// =====================================================================
extern "C" void kernel_launch(void* const* d_in, const int* in_sizes, int n_in,
                              void* d_out, int out_size, void* d_ws, size_t ws_size,
                              hipStream_t stream) {
    const float* x    = (const float*)d_in[0];   // (2,128,256,256)
    const float* Wqkv = (const float*)d_in[1];   // (384,128)
    const float* Wout = (const float*)d_in[2];   // (128,128)
    const float* relH = (const float*)d_in[3];   // (23,32)
    const float* relW = (const float*)d_in[4];   // (23,32)

    __bf16* qkv  = (__bf16*)d_ws;
    __bf16* obuf = (__bf16*)((char*)d_ws + (size_t)2 * 384 * HW * sizeof(__bf16));
    float*  out  = (float*)d_out;

    qkv_gemm   <<<dim3(HW / 64, 3, 2), 256, 0, stream>>>(x, Wqkv, qkv);
    attn_kernel<<<dim3(2 * NWIN * NWIN * HEADS), 128, 0, stream>>>(qkv, relH, relW, obuf);
    out_gemm   <<<dim3(HW / 64, 1, 2), 256, 0, stream>>>(obuf, Wout, out);
}
